// Lfm2VlMultiModalProjector_82892868813521
// MI455X (gfx1250) — matrix operand order repaired and checked by reading, not compiled, not run on hardware
//
#include <hip/hip_runtime.h>
#include <hip/hip_bf16.h>

// ---------------------------------------------------------------------------
// MI455X (gfx1250) LFM2-VL multimodal projector:
//   unshuffle(gather) -> LayerNorm -> [GEMM1 + GELU] -> [GEMM2 + bias]
// Compute-bound (735 GFLOP vs ~460MB HBM => ~20us mem at 23.3TB/s, compute
// dominates), so both GEMMs run on v_wmma_f32_16x16x32_bf16 (wave32 WMMA)
// with fp32 accumulate. Weights are pre-transposed to [N][K] bf16 so GEMM
// tile staging is pure contiguous async DMA (global_load_async_to_lds_b128)
// into double-buffered LDS, overlapping HBM traffic with the matrix pipes.
// ---------------------------------------------------------------------------

typedef __bf16 bf16_t;
typedef __attribute__((ext_vector_type(16))) __bf16 v16bf;
typedef __attribute__((ext_vector_type(8)))  float  v8f;
typedef __attribute__((ext_vector_type(4)))  int    v4i_t;

typedef __attribute__((address_space(1))) v4i_t* gv4i_ptr;   // global int4*
typedef __attribute__((address_space(3))) v4i_t* lv4i_ptr;   // LDS    int4*

#define FACTOR   2
#define H_VIS    1024
#define IC       4096     // H_VIS * FACTOR^2
#define PH       4096     // projector hidden
#define TH_OUT   2048     // text hidden
#define N_OUT    14592    // 58368 / 4   (== 114 * 128)
#define NTILES   30
#define LN_EPSF  1e-5f

static __device__ __forceinline__ bf16_t f32_to_bf16(float f) {
    unsigned u = __float_as_uint(f);
    u += 0x7FFFu + ((u >> 16) & 1u);           // round-to-nearest-even
    unsigned short s = (unsigned short)(u >> 16);
    return __builtin_bit_cast(bf16_t, s);
}

static __device__ __forceinline__ void store_out(float* p, float v)  { *p = v; }
static __device__ __forceinline__ void store_out(bf16_t* p, float v) { *p = f32_to_bf16(v); }

union FragBF { uint4 q[2]; v16bf v; };        // 32 bytes = 8 VGPRs

// ---- CDNA5 async global->LDS copy (16B per lane), ASYNCcnt-tracked --------
static __device__ __forceinline__ void async_copy_b128(void* lds_dst, const void* gsrc) {
#if __has_builtin(__builtin_amdgcn_global_load_async_to_lds_b128)
    __builtin_amdgcn_global_load_async_to_lds_b128(
        (gv4i_ptr)gsrc, (lv4i_ptr)lds_dst, 0, 0);
#else
    unsigned loff = (unsigned)(unsigned long long)
        (__attribute__((address_space(3))) void*)lds_dst;
    unsigned long long ga = (unsigned long long)gsrc;
    asm volatile("global_load_async_to_lds_b128 %0, %1, off"
                 :: "v"(loff), "v"(ga) : "memory");
#endif
}

static __device__ __forceinline__ void wait_async0() {
#if __has_builtin(__builtin_amdgcn_s_wait_asynccnt)
    __builtin_amdgcn_s_wait_asynccnt(0);
#else
    asm volatile("s_wait_asynccnt 0x0" ::: "memory");
#endif
}

// ---------------------------------------------------------------------------
// fp32 (K x N row-major) -> bf16 transposed (N x K row-major), 64x64 LDS tile.
// Pays the B-transpose once so GEMM staging needs no per-element shuffles.
// ---------------------------------------------------------------------------
__global__ void __launch_bounds__(256)
lfm2vl_cvt_transpose_kernel(const float* __restrict__ in,   // K x N
                            bf16_t* __restrict__ out,       // N x K
                            int K, int N) {
    __shared__ __align__(16) bf16_t t[64][72];              // +8 pad
    const int tid = threadIdx.x;
    const int n0  = blockIdx.x * 64;
    const int k0  = blockIdx.y * 64;

#pragma unroll
    for (int i = 0; i < 4; ++i) {
        int lin = tid + i * 256;          // 0..1023
        int kk  = lin >> 4;               // 0..63
        int f4  = lin & 15;               // 16 float4 per 64-wide row
        float4 v = *(const float4*)(in + (size_t)(k0 + kk) * N + n0 + f4 * 4);
        t[f4 * 4 + 0][kk] = f32_to_bf16(v.x);
        t[f4 * 4 + 1][kk] = f32_to_bf16(v.y);
        t[f4 * 4 + 2][kk] = f32_to_bf16(v.z);
        t[f4 * 4 + 3][kk] = f32_to_bf16(v.w);
    }
    __syncthreads();
#pragma unroll
    for (int i = 0; i < 2; ++i) {
        int lin = tid + i * 256;          // 0..511
        int nn  = lin >> 3;               // 0..63
        int c8  = lin & 7;                // 8 bf16 chunks per 64-wide row
        *(uint4*)(out + (size_t)(n0 + nn) * K + k0 + c8 * 8) =
            *(const uint4*)&t[nn][c8 * 8];
    }
}

// ---------------------------------------------------------------------------
// Fused pixel-unshuffle gather + LayerNorm + bf16 cast.
// One 256-thread block per output row (4096 features = 16 per thread).
// ---------------------------------------------------------------------------
__global__ void __launch_bounds__(256)
lfm2vl_unshuffle_ln_kernel(const float* __restrict__ vf,
                           const int*   __restrict__ ss, int ssStride,
                           const float* __restrict__ gamma,
                           const float* __restrict__ beta,
                           bf16_t* __restrict__ xout) {
    const int r = blockIdx.x;

    // Locate tile for this output row (uniform 30-step scan).
    int off_in = 0, off_out = 0, w = 2, rl = 0;
    for (int t = 0; t < NTILES; ++t) {
        int th = ss[(2 * t)     * ssStride];
        int tw = ss[(2 * t + 1) * ssStride];
        int orows = (th / FACTOR) * (tw / FACTOR);
        if (r < off_out + orows) { w = tw; rl = r - off_out; break; }
        off_out += orows;
        off_in  += th * tw;
    }
    const int wb = w / FACTOR;
    const int bh = rl / wb, bw = rl % wb;

    const int tid   = threadIdx.x;
    const int cbase = tid * 16;            // 16 contiguous features, same segment
    const int seg   = cbase / H_VIS;       // which of the 4 gathered rows
    const int c0    = cbase % H_VIS;
    const int irow  = off_in + (FACTOR * bh + (seg >> 1)) * w + FACTOR * bw + (seg & 1);

    const float* src = vf + (size_t)irow * H_VIS + c0;
    float v[16];
    float s = 0.f, sq = 0.f;
#pragma unroll
    for (int j = 0; j < 4; ++j) {
        float4 f = *(const float4*)(src + 4 * j);
        v[4*j+0] = f.x; v[4*j+1] = f.y; v[4*j+2] = f.z; v[4*j+3] = f.w;
    }
#pragma unroll
    for (int j = 0; j < 16; ++j) { s += v[j]; sq += v[j] * v[j]; }

    __shared__ float red0[256], red1[256];
    red0[tid] = s; red1[tid] = sq;
    __syncthreads();
    for (int st = 128; st > 0; st >>= 1) {
        if (tid < st) { red0[tid] += red0[tid + st]; red1[tid] += red1[tid + st]; }
        __syncthreads();
    }
    const float mu   = red0[0] * (1.0f / IC);
    const float var  = red1[0] * (1.0f / IC) - mu * mu;
    const float rstd = rsqrtf(var + LN_EPSF);

    bf16_t* dst = xout + (size_t)r * IC + cbase;
#pragma unroll
    for (int j = 0; j < 16; ++j) {
        int c = cbase + j;
        dst[j] = f32_to_bf16((v[j] - mu) * rstd * gamma[c] + beta[c]);
    }
}

// ---------------------------------------------------------------------------
// bf16 WMMA GEMM:  C[M,N] = act(A[M,K] @ Bt[N,K]^T + bias)
// Block tile 128x128, BK=32, 256 threads = 8 wave32 waves (4M x 2N),
// each wave: 32x64 = 2x4 tiles of v_wmma_f32_16x16x32_bf16.
// Both A and Bt tiles are [row][k] -> staged by async DMA into double-
// buffered LDS; DMA of tile k+1 overlaps the WMMAs of tile k.
// ---------------------------------------------------------------------------
template <bool GELU_ACT, typename OutT>
__global__ void __launch_bounds__(256)
lfm2vl_gemm_bf16_wmma_kernel(const bf16_t* __restrict__ A,    // M x K
                             const bf16_t* __restrict__ Bt,   // N x K
                             const float*  __restrict__ bias,
                             OutT* __restrict__ C,            // M x N
                             int M, int N, int K) {
    (void)M;
    __shared__ __align__(16) bf16_t ldsA[2][128][40];   // [buf][m][k], +8 pad
    __shared__ __align__(16) bf16_t ldsB[2][128][40];   // [buf][n][k], +8 pad

    const int tid    = threadIdx.x;
    const int lane   = tid & 31;
    const int wid    = tid >> 5;
    const int l15    = lane & 15;
    const int hi     = (lane >> 4) & 1;
    const int mBlock = blockIdx.y * 128;
    const int nBlock = blockIdx.x * 128;
    const int waveM  = (wid & 3) * 32;
    const int waveN  = (wid >> 2) * 64;

    // Per-thread staging coordinates (each thread DMAs 2x16B per tile).
    const int r0 = tid >> 2;              // row for chunk 0   (0..63)
    const int r1 = (tid + 256) >> 2;      // row for chunk 1   (64..127)
    const int c0 = (tid & 3) * 8;         // k-offset within tile

    v8f acc[2][4] = {};

    const int KT = K >> 5;                // K / 32 iterations

    // Prologue: DMA tile 0 into buffer 0.
    {
        async_copy_b128(&ldsA[0][r0][c0], A  + (size_t)(mBlock + r0) * K + c0);
        async_copy_b128(&ldsA[0][r1][c0], A  + (size_t)(mBlock + r1) * K + c0);
        async_copy_b128(&ldsB[0][r0][c0], Bt + (size_t)(nBlock + r0) * K + c0);
        async_copy_b128(&ldsB[0][r1][c0], Bt + (size_t)(nBlock + r1) * K + c0);
        wait_async0();
        __syncthreads();
    }

    for (int kt = 0; kt < KT; ++kt) {
        const int cur = kt & 1;
        // Kick off DMA for the next K-tile into the other buffer.
        if (kt + 1 < KT) {
            const int nxt = cur ^ 1;
            const int kb  = (kt + 1) << 5;
            async_copy_b128(&ldsA[nxt][r0][c0], A  + (size_t)(mBlock + r0) * K + kb + c0);
            async_copy_b128(&ldsA[nxt][r1][c0], A  + (size_t)(mBlock + r1) * K + kb + c0);
            async_copy_b128(&ldsB[nxt][r0][c0], Bt + (size_t)(nBlock + r0) * K + kb + c0);
            async_copy_b128(&ldsB[nxt][r1][c0], Bt + (size_t)(nBlock + r1) * K + kb + c0);
        }

        // A fragments: lane<16 holds K{0..7,16..23}, lane>=16 K{8..15,24..31}
        FragBF aFrag[2];
        const int ka = hi ? 8 : 0;
#pragma unroll
        for (int mi = 0; mi < 2; ++mi) {
            int row = waveM + mi * 16 + l15;
            aFrag[mi].q[0] = *(const uint4*)&ldsA[cur][row][ka];
            aFrag[mi].q[1] = *(const uint4*)&ldsA[cur][row][ka + 16];
        }
        // B fragments: lane<16 holds K{0..7,16..23} of column n, lane>=16 K{8..15,24..31}
        const int ko = hi ? 8 : 0;
#pragma unroll
        for (int ni = 0; ni < 4; ++ni) {
            int col = waveN + ni * 16 + l15;
            FragBF bFrag;
            bFrag.q[0] = *(const uint4*)&ldsB[cur][col][ko];
            bFrag.q[1] = *(const uint4*)&ldsB[cur][col][ko + 16];
#pragma unroll
            for (int mi = 0; mi < 2; ++mi) {
                acc[mi][ni] = __builtin_amdgcn_wmma_f32_16x16x32_bf16(
                    false, aFrag[mi].v, false, bFrag.v,
                    (short)0, acc[mi][ni], false, false);
            }
        }

        wait_async0();      // next tile's DMA landed
        __syncthreads();    // everyone done reading cur, next buf valid
    }

    // Epilogue: C/D layout -> elem e of lane L is (M = e + 8*hi, N = L&15)
#pragma unroll
    for (int mi = 0; mi < 2; ++mi) {
#pragma unroll
        for (int ni = 0; ni < 4; ++ni) {
            int n = nBlock + waveN + ni * 16 + l15;
            float bn = bias[n];
#pragma unroll
            for (int e = 0; e < 8; ++e) {
                int m = mBlock + waveM + mi * 16 + e + hi * 8;
                float val = acc[mi][ni][e] + bn;
                if (GELU_ACT)
                    val = 0.5f * val * (1.0f + erff(val * 0.7071067811865475f));
                store_out(C + (size_t)m * N + n, val);
            }
        }
    }
}

// ---------------------------------------------------------------------------
extern "C" void kernel_launch(void* const* d_in, const int* in_sizes, int n_in,
                              void* d_out, int out_size, void* d_ws, size_t ws_size,
                              hipStream_t stream) {
    (void)n_in; (void)out_size; (void)ws_size;
    const float* vf    = (const float*)d_in[0];
    const int*   ss    = (const int*)  d_in[1];
    const float* gamma = (const float*)d_in[2];
    const float* beta  = (const float*)d_in[3];
    const float* w1    = (const float*)d_in[4];
    const float* b1    = (const float*)d_in[5];
    const float* w2    = (const float*)d_in[6];
    const float* b2    = (const float*)d_in[7];
    float* out = (float*)d_out;

    // spatial_shapes may arrive as int32 (60 elems) or int64 (2 dwords each)
    const int ssStride = (in_sizes[1] == 2 * NTILES) ? 1 : 2;

    // workspace layout (~289 MB)
    char* ws = (char*)d_ws;
    const size_t szX  = (size_t)N_OUT * IC * sizeof(bf16_t);
    const size_t szH  = (size_t)N_OUT * PH * sizeof(bf16_t);
    const size_t szW1 = (size_t)IC    * PH * sizeof(bf16_t);
    bf16_t* xb   = (bf16_t*)ws;
    bf16_t* hb   = (bf16_t*)(ws + szX);
    bf16_t* w1t  = (bf16_t*)(ws + szX + szH);              // PH x IC  (N x K)
    bf16_t* w2t  = (bf16_t*)(ws + szX + szH + szW1);       // TH x PH  (N x K)

    // 1) weights fp32 -> bf16, transposed to [N][K]
    lfm2vl_cvt_transpose_kernel<<<dim3(PH / 64, IC / 64), 256, 0, stream>>>(
        w1, w1t, IC, PH);
    lfm2vl_cvt_transpose_kernel<<<dim3(TH_OUT / 64, PH / 64), 256, 0, stream>>>(
        w2, w2t, PH, TH_OUT);

    // 2) gather + LN -> x_bf16 (14592 x 4096)
    lfm2vl_unshuffle_ln_kernel<<<N_OUT, 256, 0, stream>>>(vf, ss, ssStride,
                                                          gamma, beta, xb);
    // 3) GEMM1 + GELU -> h_bf16 (14592 x 4096)
    lfm2vl_gemm_bf16_wmma_kernel<true, bf16_t>
        <<<dim3(PH / 128, N_OUT / 128), 256, 0, stream>>>(xb, w1t, b1, hb,
                                                          N_OUT, PH, IC);
    // 4) GEMM2 + bias -> out fp32 (14592 x 2048)
    lfm2vl_gemm_bf16_wmma_kernel<false, float>
        <<<dim3(TH_OUT / 128, N_OUT / 128), 256, 0, stream>>>(hb, w2t, b2, out,
                                                              N_OUT, TH_OUT, PH);
}
